// MultiHeadLatentAttention_80771154968634
// MI455X (gfx1250) — matrix operand order repaired and checked by reading, not compiled, not run on hardware
//
#include <hip/hip_runtime.h>
#include <hip/hip_bf16.h>
#include <math.h>

typedef __attribute__((ext_vector_type(8)))  float  v8f;
typedef __attribute__((ext_vector_type(8)))  __bf16 v8bf;
typedef __attribute__((ext_vector_type(16))) __bf16 v16bf;

namespace {
constexpr int Bc = 2, Sc = 1024, HIDc = 2048;
constexpr int Hc = 16, Dc = 128, Rc = 64;
constexpr int KVCc = 512, QCc = 1536;
}

// Pack two 8-element bf16 chunks into one 16-element WMMA fragment.
__device__ inline v16bf pack16(v8bf c0, v8bf c1) {
  v16bf r;
#pragma unroll
  for (int i = 0; i < 8; ++i) { r[i] = c0[i]; r[i + 8] = c1[i]; }
  return r;
}

// Split fp32 into bf16 hi/lo (hi RNE; lo = residual). hi+lo carries ~16 mantissa bits.
__device__ inline void split2(float x, __bf16& h, __bf16& l) {
  h = (__bf16)x;
  l = (__bf16)(x - (float)h);
}

// ---------------------------------------------------------------------------
// C[M,N] = A[M,K] @ W[N,K]^T + bias[N], from PRE-SPLIT bf16 hi/lo operands.
// Block 128 threads (4 waves). Block tile 128x64, wave tile 32x64, K-step 32.
// Double-buffered LDS (48 KB): issue next-slab global loads, run 24
// v_wmma_f32_16x16x32_bf16 on the current buffer while loads are in flight,
// then store staged registers. bf16x3 (hh+hl+lh) ~1e-5 relative vs fp32.
// ---------------------------------------------------------------------------
__global__ __launch_bounds__(128) void gemm_bias_pre(
    const __bf16* __restrict__ Ahg, const __bf16* __restrict__ Alg,
    const __bf16* __restrict__ Whg, const __bf16* __restrict__ Wlg,
    const float* __restrict__ bias, float* __restrict__ C,
    int M, int N, int K)
{
  __shared__ __bf16 AhL[2][128 * 32];
  __shared__ __bf16 AlL[2][128 * 32];
  __shared__ __bf16 WhL[2][64 * 32];
  __shared__ __bf16 WlL[2][64 * 32];
  const int tid  = threadIdx.x;
  const int wave = tid >> 5;
  const int lane = tid & 31;
  const int half = lane >> 4;
  const int l16  = lane & 15;
  const int mBase = blockIdx.y * 128;
  const int nBase = blockIdx.x * 64;

  v8f acc[2][4];
#pragma unroll
  for (int i = 0; i < 2; ++i)
#pragma unroll
    for (int nc = 0; nc < 4; ++nc)
#pragma unroll
      for (int j = 0; j < 8; ++j) acc[i][nc][j] = 0.0f;

  // Staging registers: A row (tid) 32 bf16 hi+lo; W half-row 16 bf16 hi+lo.
  v8bf ra[8];
  v8bf rw[4];
  const size_t arowg = (size_t)(mBase + tid) * K;               // A row per thread
  const size_t wrowg = (size_t)(nBase + (tid >> 1)) * K + (tid & 1) * 16;

  auto gload = [&](int k0) {
    const v8bf* ah = (const v8bf*)(Ahg + arowg + k0);
    const v8bf* al = (const v8bf*)(Alg + arowg + k0);
    const v8bf* wh = (const v8bf*)(Whg + wrowg + k0);
    const v8bf* wl = (const v8bf*)(Wlg + wrowg + k0);
#pragma unroll
    for (int i = 0; i < 4; ++i) { ra[i] = ah[i]; ra[4 + i] = al[i]; }
    rw[0] = wh[0]; rw[1] = wh[1];
    rw[2] = wl[0]; rw[3] = wl[1];
  };
  auto lstore = [&](int buf) {
    v8bf* ah = (v8bf*)&AhL[buf][tid * 32];
    v8bf* al = (v8bf*)&AlL[buf][tid * 32];
    v8bf* wh = (v8bf*)&WhL[buf][(tid >> 1) * 32 + (tid & 1) * 16];
    v8bf* wl = (v8bf*)&WlL[buf][(tid >> 1) * 32 + (tid & 1) * 16];
#pragma unroll
    for (int i = 0; i < 4; ++i) { ah[i] = ra[i]; al[i] = ra[4 + i]; }
    wh[0] = rw[0]; wh[1] = rw[1];
    wl[0] = rw[2]; wl[1] = rw[3];
  };

  gload(0);
  lstore(0);
  int buf = 0;

  for (int k0 = 0; k0 < K; k0 += 32) {
    __syncthreads();
    const bool more = (k0 + 32) < K;
    if (more) {
      gload(k0 + 32);                      // in flight during WMMA below
      if (k0 + 64 < K) {                   // prefetch the slab after that
        __builtin_prefetch(Ahg + arowg + k0 + 64, 0, 2);
        __builtin_prefetch(Whg + wrowg + k0 + 64, 0, 2);
      }
    }

    // A frags (16x32 bf16): lane = M row; elems [0..7]=K(8h..), [8..15]=K(16+8h..)
    v16bf afh[2], afl[2];
#pragma unroll
    for (int i = 0; i < 2; ++i) {
      const int arow = (wave * 32 + i * 16 + l16) * 32;
      afh[i] = pack16(*(const v8bf*)&AhL[buf][arow + 8 * half],
                      *(const v8bf*)&AhL[buf][arow + 16 + 8 * half]);
      afl[i] = pack16(*(const v8bf*)&AlL[buf][arow + 8 * half],
                      *(const v8bf*)&AlL[buf][arow + 16 + 8 * half]);
    }
#pragma unroll
    for (int nc = 0; nc < 4; ++nc) {
      // B frag (32x16 bf16): lane = N col; elems [0..7]=K(8h..), [8..15]=K(16+8h..); B[k][n]=W[n][k]
      const int wrow = (nc * 16 + l16) * 32 + 8 * half;
      v16bf wfh = pack16(*(const v8bf*)&WhL[buf][wrow],
                         *(const v8bf*)&WhL[buf][wrow + 16]);
      v16bf wfl = pack16(*(const v8bf*)&WlL[buf][wrow],
                         *(const v8bf*)&WlL[buf][wrow + 16]);
#pragma unroll
      for (int i = 0; i < 2; ++i) {
        acc[i][nc] = __builtin_amdgcn_wmma_f32_16x16x32_bf16(
            false, afh[i], false, wfl, (short)0, acc[i][nc], false, false);
        acc[i][nc] = __builtin_amdgcn_wmma_f32_16x16x32_bf16(
            false, afl[i], false, wfh, (short)0, acc[i][nc], false, false);
        acc[i][nc] = __builtin_amdgcn_wmma_f32_16x16x32_bf16(
            false, afh[i], false, wfh, (short)0, acc[i][nc], false, false);
      }
    }
    if (more) lstore(buf ^ 1);             // waits loads; fills other buffer
    buf ^= 1;
  }

#pragma unroll
  for (int nc = 0; nc < 4; ++nc) {
    int col = nBase + nc * 16 + l16;
    float bv = bias[col];
#pragma unroll
    for (int i = 0; i < 2; ++i)
#pragma unroll
      for (int r = 0; r < 8; ++r) {
        int row = mBase + wave * 32 + i * 16 + r + 8 * half;
        C[(size_t)row * N + col] = acc[i][nc][r] + bv;
      }
  }
}

// ---------------------------------------------------------------------------
// Vectorized fp32 -> bf16 hi/lo split (4 elements/thread; n must be mult of 4).
// ---------------------------------------------------------------------------
__global__ __launch_bounds__(256) void split_bf16x4(
    const float* __restrict__ x, __bf16* __restrict__ h,
    __bf16* __restrict__ l, int n4)
{
  int i = blockIdx.x * 256 + threadIdx.x;
  if (i >= n4) return;
  float4 v = ((const float4*)x)[i];
  int o = i * 4;
  split2(v.x, h[o + 0], l[o + 0]);
  split2(v.y, h[o + 1], l[o + 1]);
  split2(v.z, h[o + 2], l[o + 2]);
  split2(v.w, h[o + 3], l[o + 3]);
}

// V: [B,S,H,D] fp32  ->  Vt hi/lo: [B,H,D,S] bf16 (transposed for PV B-frags)
__global__ __launch_bounds__(256) void vsplit_t(
    const float* __restrict__ v, __bf16* __restrict__ th,
    __bf16* __restrict__ tl)
{
  int i = blockIdx.x * 256 + threadIdx.x;   // over B*H*D*S
  int s  = i & (Sc - 1);
  int t  = i >> 10;
  int d  = t & (Dc - 1);
  int bh = t >> 7;
  int b  = bh >> 4, h = bh & (Hc - 1);
  float x = v[((size_t)(b * Sc + s) * Hc + h) * Dc + d];
  split2(x, th[i], tl[i]);
}

// ---------------------------------------------------------------------------
// Flash attention, bf16x3 WMMA. One wave = 16 query rows of one (b,h);
// key tile = 32 so PV gets full K=32 WMMAs. Q/K from pre-split [B,S,H,D]
// bf16 hi/lo; V from pre-split transposed [B,H,D,S] bf16 hi/lo.
// ---------------------------------------------------------------------------
__global__ __launch_bounds__(128) void mla_flash_bf16x3(
    const __bf16* __restrict__ Qh, const __bf16* __restrict__ Ql,
    const __bf16* __restrict__ Kh, const __bf16* __restrict__ Kl,
    const __bf16* __restrict__ Vth, const __bf16* __restrict__ Vtl,
    float* __restrict__ Oa)
{
  __shared__ __bf16 Pph[4][16 * 32];
  __shared__ __bf16 Ppl[4][16 * 32];
  const int bh   = blockIdx.x;            // 0..B*H-1
  const int b    = bh / Hc, h = bh % Hc;
  const int wave = threadIdx.x >> 5;
  const int lane = threadIdx.x & 31;
  const int half = lane >> 4;
  const int l16  = lane & 15;
  const int qBase = (blockIdx.y * 4 + wave) * 16;
  const size_t rs = (size_t)Hc * Dc;
  const __bf16* qph = Qh + ((size_t)b * Sc * Hc + h) * Dc;
  const __bf16* qpl = Ql + ((size_t)b * Sc * Hc + h) * Dc;
  const __bf16* kph = Kh + ((size_t)b * Sc * Hc + h) * Dc;
  const __bf16* kpl = Kl + ((size_t)b * Sc * Hc + h) * Dc;
  const __bf16* vth = Vth + (size_t)bh * Dc * Sc;
  const __bf16* vtl = Vtl + (size_t)bh * Dc * Sc;
  __bf16* plh = Pph[wave];
  __bf16* pll = Ppl[wave];
  const float scale = 0.08838834764831845f;   // 1/sqrt(128)

  // Preload Q 16x128 as 4 K-step fragments (hi/lo)
  v16bf qah[4], qal[4];
#pragma unroll
  for (int st = 0; st < 4; ++st) {
    const __bf16* qb0 = qph + (size_t)(qBase + l16) * rs + st * 32;
    const __bf16* qb1 = qpl + (size_t)(qBase + l16) * rs + st * 32;
    qah[st] = pack16(*(const v8bf*)(qb0 + 8 * half),
                     *(const v8bf*)(qb0 + 16 + 8 * half));
    qal[st] = pack16(*(const v8bf*)(qb1 + 8 * half),
                     *(const v8bf*)(qb1 + 16 + 8 * half));
  }

  float m_r[8], l_r[8];
  v8f oacc[8];
#pragma unroll
  for (int r = 0; r < 8; ++r) { m_r[r] = -1e30f; l_r[r] = 0.0f; }
#pragma unroll
  for (int dc = 0; dc < 8; ++dc)
#pragma unroll
    for (int r = 0; r < 8; ++r) oacc[dc][r] = 0.0f;

  for (int kt = 0; kt < Sc / 32; ++kt) {
    const int keyBase = kt * 32;

    // S tile 16x32 = Q K^T : 2 key groups x 4 K-steps x 3 terms = 24 WMMA
    v8f sacc[2];
#pragma unroll
    for (int g = 0; g < 2; ++g)
#pragma unroll
      for (int r = 0; r < 8; ++r) sacc[g][r] = 0.0f;
#pragma unroll
    for (int st = 0; st < 4; ++st) {
#pragma unroll
      for (int g = 0; g < 2; ++g) {
        const __bf16* kb0 = kph + (size_t)(keyBase + g * 16 + l16) * rs + st * 32 + 8 * half;
        const __bf16* kb1 = kpl + (size_t)(keyBase + g * 16 + l16) * rs + st * 32 + 8 * half;
        v16bf kfh = pack16(*(const v8bf*)kb0, *(const v8bf*)(kb0 + 16));
        v16bf kfl = pack16(*(const v8bf*)kb1, *(const v8bf*)(kb1 + 16));
        sacc[g] = __builtin_amdgcn_wmma_f32_16x16x32_bf16(
            false, qah[st], false, kfl, (short)0, sacc[g], false, false);
        sacc[g] = __builtin_amdgcn_wmma_f32_16x16x32_bf16(
            false, qal[st], false, kfh, (short)0, sacc[g], false, false);
        sacc[g] = __builtin_amdgcn_wmma_f32_16x16x32_bf16(
            false, qah[st], false, kfh, (short)0, sacc[g], false, false);
      }
    }

    // Online softmax over 32 columns (two 16-wide score accs)
#pragma unroll
    for (int r = 0; r < 8; ++r) {
      float x0 = sacc[0][r] * scale;
      float x1 = sacc[1][r] * scale;
      float mx = fmaxf(x0, x1);
      mx = fmaxf(mx, __shfl_xor(mx, 1, 32));
      mx = fmaxf(mx, __shfl_xor(mx, 2, 32));
      mx = fmaxf(mx, __shfl_xor(mx, 4, 32));
      mx = fmaxf(mx, __shfl_xor(mx, 8, 32));
      float mn = fmaxf(m_r[r], mx);
      float p0 = __expf(x0 - mn);
      float p1 = __expf(x1 - mn);
      float sum = p0 + p1;
      sum += __shfl_xor(sum, 1, 32);
      sum += __shfl_xor(sum, 2, 32);
      sum += __shfl_xor(sum, 4, 32);
      sum += __shfl_xor(sum, 8, 32);
      float al = __expf(m_r[r] - mn);
      l_r[r] = l_r[r] * al + sum;
      m_r[r] = mn;
#pragma unroll
      for (int dc = 0; dc < 8; ++dc) oacc[dc][r] *= al;
      int row = r + 8 * half;
      split2(p0, plh[row * 32 + l16],      pll[row * 32 + l16]);
      split2(p1, plh[row * 32 + 16 + l16], pll[row * 32 + 16 + l16]);
    }
    __syncthreads();

    // O += P V : A = P 16x32 (hi/lo from LDS), B = Vt columns (contig bf16)
    const __bf16* pb = plh + l16 * 32;
    const __bf16* qb = pll + l16 * 32;
    v16bf pfh = pack16(*(const v8bf*)(pb + 8 * half),
                       *(const v8bf*)(pb + 16 + 8 * half));
    v16bf pfl = pack16(*(const v8bf*)(qb + 8 * half),
                       *(const v8bf*)(qb + 16 + 8 * half));
#pragma unroll
    for (int dc = 0; dc < 8; ++dc) {
      const __bf16* vb0 = vth + (size_t)(dc * 16 + l16) * Sc + keyBase + 8 * half;
      const __bf16* vb1 = vtl + (size_t)(dc * 16 + l16) * Sc + keyBase + 8 * half;
      v16bf vfh = pack16(*(const v8bf*)vb0, *(const v8bf*)(vb0 + 16));
      v16bf vfl = pack16(*(const v8bf*)vb1, *(const v8bf*)(vb1 + 16));
      oacc[dc] = __builtin_amdgcn_wmma_f32_16x16x32_bf16(
          false, pfh, false, vfl, (short)0, oacc[dc], false, false);
      oacc[dc] = __builtin_amdgcn_wmma_f32_16x16x32_bf16(
          false, pfl, false, vfh, (short)0, oacc[dc], false, false);
      oacc[dc] = __builtin_amdgcn_wmma_f32_16x16x32_bf16(
          false, pfh, false, vfh, (short)0, oacc[dc], false, false);
    }
    __syncthreads();
  }

#pragma unroll
  for (int r = 0; r < 8; ++r) {
    float inv = 1.0f / l_r[r];
    int row = qBase + r + 8 * half;
    float* orow = Oa + ((size_t)(b * Sc + row) * Hc + h) * Dc;
#pragma unroll
    for (int dc = 0; dc < 8; ++dc)
      orow[dc * 16 + l16] = oacc[dc][r] * inv;
  }
}

// ---------------------------------------------------------------------------
// k_r = RoPE( k @ W_KR^T + b_KR ) from exact fp32 k — tiny, VALU kernel.
// ---------------------------------------------------------------------------
__global__ __launch_bounds__(256) void kr_rope(
    const float* __restrict__ Kt, const float* __restrict__ Wkr,
    const float* __restrict__ bkr, float* __restrict__ KRo)
{
  int idx = blockIdx.x * blockDim.x + threadIdx.x;
  if (idx >= Bc * Sc * Hc * 32) return;
  int i = idx & 31;
  int m = idx >> 5;
  int s = (m / Hc) % Sc;
  const float* krow = Kt + (size_t)m * Dc;
  const float* w0 = Wkr + (size_t)i * Dc;
  const float* w1 = Wkr + (size_t)(i + 32) * Dc;
  float d0 = bkr[i], d1 = bkr[i + 32];
  for (int d = 0; d < Dc; ++d) {
    float kv = krow[d];
    d0 += kv * w0[d];
    d1 += kv * w1[d];
  }
  float fr  = powf(10000.0f, -(float)i / 32.0f);
  float ang = (float)s * fr;
  float cs = cosf(ang), sn = sinf(ang);
  KRo[(size_t)m * Rc + i]      = d0 * cs - d1 * sn;
  KRo[(size_t)m * Rc + i + 32] = d1 * cs + d0 * sn;
}

// ---------------------------------------------------------------------------
extern "C" void kernel_launch(void* const* d_in, const int* in_sizes, int n_in,
                              void* d_out, int out_size, void* d_ws, size_t ws_size,
                              hipStream_t stream)
{
  (void)in_sizes; (void)n_in; (void)out_size; (void)ws_size;

  const float* hidden = (const float*)d_in[0];
  const float* Wdkv = (const float*)d_in[1];
  const float* bdkv = (const float*)d_in[2];
  const float* Wuk  = (const float*)d_in[3];
  const float* buk  = (const float*)d_in[4];
  const float* Wuv  = (const float*)d_in[5];
  const float* buv  = (const float*)d_in[6];
  const float* Wkr  = (const float*)d_in[7];
  const float* bkr  = (const float*)d_in[8];
  const float* Wdq  = (const float*)d_in[9];
  const float* bdq  = (const float*)d_in[10];
  const float* Wuq  = (const float*)d_in[11];
  const float* buq  = (const float*)d_in[12];
  // d_in[13], d_in[14] = W_QR_w/b: dead in the reference (q_r never returned/used)
  const float* Wo   = (const float*)d_in[15];
  const float* bo   = (const float*)d_in[16];

  // d_out = [output | c_kv | k_r] flat fp32
  float* out = (float*)d_out;
  float* ckv = out + (size_t)Bc * Sc * HIDc;
  float* kr  = ckv + (size_t)Bc * Sc * KVCc;

  const size_t N1M = 1u << 20;
  const size_t N4M = 1u << 22;        // B*S*H*D elements
  float* ws   = (float*)d_ws;
  float* kbuf = ws;                   // fp32 k   (4M f32)
  float* vbuf = kbuf + N4M;           // fp32 v   (4M)
  float* cq   = vbuf + N4M;           // fp32 c_q (3M)
  float* qbuf = cq + 3 * N1M;         // fp32 q   (4M); reused as attn after split
  float* attn = qbuf;
  __bf16* hid_h = (__bf16*)(qbuf + N4M);   // hidden split (4M pair)
  __bf16* hid_l = hid_h + N4M;
  __bf16* wsc_h = hid_l + N4M;        // current-weight scratch (<= 4M pair)
  __bf16* wsc_l = wsc_h + N4M;
  __bf16* ckv_h = wsc_l + N4M;        // c_kv split (1M pair)
  __bf16* ckv_l = ckv_h + N1M;
  __bf16* cq_h  = ckv_l + N1M;        // c_q split (3M pair)
  __bf16* cq_l  = cq_h + 3 * N1M;
  __bf16* qh    = cq_l + 3 * N1M;     // q/k/vT splits (4M pairs each)
  __bf16* ql    = qh + N4M;
  __bf16* kh    = ql + N4M;
  __bf16* kl    = kh + N4M;
  __bf16* vth   = kl + N4M;
  __bf16* vtl   = vth + N4M;
  __bf16* at_h  = vtl + N4M;          // attn split (4M pair)
  __bf16* at_l  = at_h + N4M;

  const int M = Bc * Sc;   // 2048
  dim3 blk(128);

  auto SPLIT = [&](const float* src, __bf16* h, __bf16* l, size_t n) {
    hipLaunchKernelGGL(split_bf16x4, dim3((unsigned)(n / 1024)), dim3(256), 0,
                       stream, src, h, l, (int)(n / 4));
  };
  auto GEMM = [&](const __bf16* ah, const __bf16* al, const __bf16* wh,
                  const __bf16* wl, const float* bias, float* c, int n, int k) {
    hipLaunchKernelGGL(gemm_bias_pre, dim3(n / 64, M / 128), blk, 0, stream,
                       ah, al, wh, wl, bias, c, M, n, k);
  };

  // --- projections (weights split JIT into shared scratch; stream-ordered) ---
  SPLIT(hidden, hid_h, hid_l, (size_t)M * HIDc);
  SPLIT(Wdkv, wsc_h, wsc_l, (size_t)KVCc * HIDc);
  GEMM(hid_h, hid_l, wsc_h, wsc_l, bdkv, ckv, KVCc, HIDc);       // c_kv -> d_out
  SPLIT(ckv, ckv_h, ckv_l, (size_t)M * KVCc);
  SPLIT(Wuk, wsc_h, wsc_l, (size_t)(Hc * Dc) * KVCc);
  GEMM(ckv_h, ckv_l, wsc_h, wsc_l, buk, kbuf, Hc * Dc, KVCc);    // k
  SPLIT(Wuv, wsc_h, wsc_l, (size_t)(Hc * Dc) * KVCc);
  GEMM(ckv_h, ckv_l, wsc_h, wsc_l, buv, vbuf, Hc * Dc, KVCc);    // v
  SPLIT(Wdq, wsc_h, wsc_l, (size_t)QCc * HIDc);
  GEMM(hid_h, hid_l, wsc_h, wsc_l, bdq, cq, QCc, HIDc);          // c_q
  SPLIT(cq, cq_h, cq_l, (size_t)M * QCc);
  SPLIT(Wuq, wsc_h, wsc_l, (size_t)(Hc * Dc) * QCc);
  GEMM(cq_h, cq_l, wsc_h, wsc_l, buq, qbuf, Hc * Dc, QCc);       // q

  // k_r = RoPE(k @ W_KR^T + b) from exact fp32 k   (d_out segment)
  hipLaunchKernelGGL(kr_rope, dim3((Bc * Sc * Hc * 32) / 256), dim3(256), 0,
                     stream, kbuf, Wkr, bkr, kr);

  // --- attention ---
  SPLIT(qbuf, qh, ql, N4M);
  SPLIT(kbuf, kh, kl, N4M);
  hipLaunchKernelGGL(vsplit_t, dim3((unsigned)(N4M / 256)), dim3(256), 0,
                     stream, vbuf, vth, vtl);
  hipLaunchKernelGGL(mla_flash_bf16x3, dim3(Bc * Hc, Sc / 64), blk, 0, stream,
                     qh, ql, kh, kl, vth, vtl, attn);

  // --- output projection ---
  SPLIT(attn, at_h, at_l, N4M);
  SPLIT(Wo, wsc_h, wsc_l, (size_t)HIDc * HIDc);
  GEMM(at_h, at_l, wsc_h, wsc_l, bo, out, HIDc, HIDc);           // output -> d_out
}
